// MyBiLSTM_687194767718
// MI455X (gfx1250) — compile-verified, matrix operand order checked by reading, not run on hardware
//
#include <hip/hip_runtime.h>
#include <hip/hip_bf16.h>

// ---------------------------------------------------------------------------
// MI455X (gfx1250) BiLSTM: bf16 WMMA for all matmuls, fp32 accumulate.
//   B=64 S=2048 D=128 H=64 G=256 K=1024
// ---------------------------------------------------------------------------

typedef __attribute__((ext_vector_type(16))) __bf16 v16bf;
typedef __attribute__((ext_vector_type(8)))  float  v8f;

union BF16Frag { unsigned int u[8]; v16bf v; };

__device__ __forceinline__ unsigned short f2bf(float f) {
    unsigned int u = __float_as_uint(f);
    u += 0x7FFFu + ((u >> 16) & 1u);   // round-to-nearest-even
    return (unsigned short)(u >> 16);
}
__device__ __forceinline__ float sigmoidf_(float x) { return 1.f / (1.f + __expf(-x)); }

// ISA 16-bit A-matrix 16x32 per-lane K map: VGPR p holds elems (2p,2p+1)
__device__ __forceinline__ int kstart(int g, int p) {
    return (p < 4) ? (g * 8 + 2 * p) : (16 + g * 8 + 2 * (p - 4));
}

// ---------------------------------------------------------------------------
// fp32 -> bf16 convert (grid-stride)
// ---------------------------------------------------------------------------
__global__ void cvt_bf16_kernel(const float* __restrict__ src,
                                unsigned short* __restrict__ dst, int n) {
    for (int i = blockIdx.x * blockDim.x + threadIdx.x; i < n;
         i += gridDim.x * blockDim.x)
        dst[i] = f2bf(src[i]);
}

__global__ void bias_sum_kernel(const float* __restrict__ a,
                                const float* __restrict__ b,
                                float* __restrict__ o, int n) {
    int i = blockIdx.x * blockDim.x + threadIdx.x;
    if (i < n) o[i] = a[i] + b[i];
}

// ---------------------------------------------------------------------------
// Gates GEMM: pre[dir][m][n] = sum_k X[m][k] * W[dir][n][k]
//   X: [M,128] bf16 row-major,  W: [2][256][128] bf16,  pre: [2][M][256] f32
// One wave computes a 16(M) x 64(N) strip (4 accumulators), K=128 in 4 WMMAs
// per tile -> 16 v_wmma_f32_16x16x32_bf16 per wave.
// ---------------------------------------------------------------------------
__global__ void gates_gemm_kernel(const unsigned short* __restrict__ Xbf,
                                  const unsigned short* __restrict__ Wbf,
                                  float* __restrict__ pre, int M) {
    int wave = (blockIdx.x * blockDim.x + threadIdx.x) >> 5;
    int lane = threadIdx.x & 31;
    int g = lane >> 4, j = lane & 15;

    int wavesPerDir = (M / 16) * 4;
    int dir = wave / wavesPerDir;
    if (dir >= 2) return;
    int rem = wave % wavesPerDir;
    int mt = rem >> 2;        // M-tile (16 rows)
    int ng = rem & 3;         // group of 4 N-tiles

    const unsigned short* W = Wbf + (size_t)dir * 256 * 128;

    v8f acc[4];
#pragma unroll
    for (int q = 0; q < 4; q++)
#pragma unroll
        for (int r = 0; r < 8; r++) acc[q][r] = 0.f;

#pragma unroll
    for (int kb = 0; kb < 4; kb++) {
        BF16Frag a;
        const unsigned short* arow = Xbf + (size_t)(mt * 16 + j) * 128 + kb * 32;
#pragma unroll
        for (int p = 0; p < 8; p++)
            a.u[p] = *(const unsigned int*)(arow + kstart(g, p));
#pragma unroll
        for (int q = 0; q < 4; q++) {
            int n = (ng * 4 + q) * 16 + j;
            BF16Frag b;
            const unsigned short* brow = W + (size_t)n * 128 + kb * 32;
#pragma unroll
            for (int p = 0; p < 8; p++)
                b.u[p] = *(const unsigned int*)(brow + kstart(g, p));
            acc[q] = __builtin_amdgcn_wmma_f32_16x16x32_bf16(
                false, a.v, false, b.v, (short)0, acc[q], false, false);
        }
    }
    float* pd = pre + (size_t)dir * M * 256;
#pragma unroll
    for (int q = 0; q < 4; q++) {
        int n = (ng * 4 + q) * 16 + j;
#pragma unroll
        for (int r = 0; r < 8; r++) {
            int m = mt * 16 + g * 8 + r;
            pd[(size_t)m * 256 + n] = acc[q][r];
        }
    }
}

// ---------------------------------------------------------------------------
// Persistent LSTM recurrence. blockIdx.x = dir + 2*chunk  (8 blocks total):
//   dir in {0(fwd),1(bwd)}, chunk = 16-batch slice. 256 threads = 8 waves.
// Per step: gates[16x256] = preGates(t) + h_t(16x64,bf16 in LDS) @ Whh^T
//   (Whh^T B-fragments live in registers for the whole kernel),
// then fused i,f,g,o sigmoid/tanh elementwise; c state in registers.
// Next step's preGates are prefetched before the barrier to hide HBM latency.
// ---------------------------------------------------------------------------
__global__ void lstm_recur_kernel(const float* __restrict__ pre,   // [2][64][S][256]
                                  const unsigned short* __restrict__ Whh, // [2][256][64]
                                  const float* __restrict__ bias,  // [2][256]
                                  const float* __restrict__ h0,    // [2][64][64]
                                  const float* __restrict__ c0,    // [2][64][64]
                                  void* __restrict__ hout,         // [64][S][128]
                                  int S, int out_bf16) {
    int dir   = blockIdx.x & 1;
    int bb0   = (blockIdx.x >> 1) * 16;
    int tid   = threadIdx.x;
    int wv    = tid >> 5;
    int lane  = tid & 31;
    int g = lane >> 4, j = lane & 15;

    __shared__ unsigned short lds_h[16 * 64];   // h_t bf16, [m][u]
    __shared__ float lds_gates[16 * 256];

    // ---- Whh^T B-fragments (constant over time) into registers -------------
    BF16Frag bfrag[2][2];                        // [ntile][kblock]
    const unsigned short* Wd = Whh + (size_t)dir * 256 * 64;
#pragma unroll
    for (int q = 0; q < 2; q++) {
        int n = wv * 32 + q * 16 + j;
#pragma unroll
        for (int kb = 0; kb < 2; kb++)
#pragma unroll
            for (int p = 0; p < 8; p++)
                bfrag[q][kb].u[p] =
                    *(const unsigned int*)(Wd + (size_t)n * 64 + kb * 32 + kstart(g, p));
    }

    // ---- per-thread elementwise ownership: 4 consecutive hidden units ------
    int m0 = (tid * 4) >> 6;
    int u0 = (tid * 4) & 63;
    const float* biasd = bias + dir * 256;
    float cst[4], bi[4], bff[4], bg[4], bo[4];
#pragma unroll
    for (int i = 0; i < 4; i++) {
        int b = bb0 + m0;
        cst[i] = c0[((size_t)dir * 64 + b) * 64 + u0 + i];
        float hv = h0[((size_t)dir * 64 + b) * 64 + u0 + i];
        lds_h[m0 * 64 + u0 + i] = f2bf(hv);
        bi[i]  = biasd[u0 + i];
        bff[i] = biasd[64 + u0 + i];
        bg[i]  = biasd[128 + u0 + i];
        bo[i]  = biasd[192 + u0 + i];
    }
    __syncthreads();

    const float* pred = pre + (size_t)dir * 64 * S * 256;
    long t = dir ? (S - 1) : 0;
    int tstep = dir ? -1 : 1;

    auto loadC = [&](v8f* dst, long tt) {
#pragma unroll
        for (int q = 0; q < 2; q++) {
            int n = wv * 32 + q * 16 + j;
#pragma unroll
            for (int r = 0; r < 8; r++) {
                int m = g * 8 + r;
                dst[q][r] = pred[((size_t)(bb0 + m) * S + tt) * 256 + n];
            }
        }
    };

    v8f cnxt[2];
    loadC(cnxt, t);

    for (int si = 0; si < S; si++) {
        v8f acc[2];
        acc[0] = cnxt[0];
        acc[1] = cnxt[1];

        // A fragments of h_t from LDS (16x64 bf16, two K=32 blocks)
        BF16Frag afr[2];
#pragma unroll
        for (int kb = 0; kb < 2; kb++)
#pragma unroll
            for (int p = 0; p < 8; p++)
                afr[kb].u[p] =
                    *(const unsigned int*)&lds_h[j * 64 + kb * 32 + kstart(g, p)];

#pragma unroll
        for (int q = 0; q < 2; q++) {
            acc[q] = __builtin_amdgcn_wmma_f32_16x16x32_bf16(
                false, afr[0].v, false, bfrag[q][0].v, (short)0, acc[q], false, false);
            acc[q] = __builtin_amdgcn_wmma_f32_16x16x32_bf16(
                false, afr[1].v, false, bfrag[q][1].v, (short)0, acc[q], false, false);
        }

        // spill gates to LDS for the elementwise re-partition
#pragma unroll
        for (int q = 0; q < 2; q++) {
            int n = wv * 32 + q * 16 + j;
#pragma unroll
            for (int r = 0; r < 8; r++)
                lds_gates[(g * 8 + r) * 256 + n] = acc[q][r];
        }

        long tn = t + tstep;
        if (si + 1 < S) loadC(cnxt, tn);   // prefetch next step (hides HBM latency)
        __syncthreads();                    // gates visible; lds_h reads done

        float hv[4];
#pragma unroll
        for (int i = 0; i < 4; i++) {
            float ig = sigmoidf_(lds_gates[m0 * 256 + u0 + i] + bi[i]);
            float fg = sigmoidf_(lds_gates[m0 * 256 + 64 + u0 + i] + bff[i]);
            float gg = tanhf(lds_gates[m0 * 256 + 128 + u0 + i] + bg[i]);
            float og = sigmoidf_(lds_gates[m0 * 256 + 192 + u0 + i] + bo[i]);
            cst[i] = fg * cst[i] + ig * gg;
            hv[i] = og * tanhf(cst[i]);
            lds_h[m0 * 64 + u0 + i] = f2bf(hv[i]);
        }
        size_t row = ((size_t)(bb0 + m0) * S + t) * 128 + dir * 64 + u0;
        if (out_bf16) {
            unsigned short* ho = (unsigned short*)hout;
#pragma unroll
            for (int i = 0; i < 4; i++) ho[row + i] = f2bf(hv[i]);
        } else {
            float* ho = (float*)hout;
#pragma unroll
            for (int i = 0; i < 4; i++) ho[row + i] = hv[i];
        }
        __syncthreads();                    // new h visible, gates free
        t = tn;
    }
}

// ---------------------------------------------------------------------------
// record[b*K+k] = h1[b, sel[b,k], :128];  out = sigmoid(record @ Wlin^T + b)
// ---------------------------------------------------------------------------
__global__ void gather_linear_kernel(const float* __restrict__ h1,
                                     const int* __restrict__ sel,
                                     const float* __restrict__ Wlin,  // [5][128]
                                     const float* __restrict__ blin,  // [5]
                                     float* __restrict__ rec,
                                     float* __restrict__ out, int S) {
    __shared__ float w[5 * 128];
    __shared__ float bl[5];
    for (int i = threadIdx.x; i < 640; i += blockDim.x) w[i] = Wlin[i];
    if (threadIdx.x < 5) bl[threadIdx.x] = blin[threadIdx.x];
    __syncthreads();

    int row = blockIdx.x * blockDim.x + threadIdx.x;
    if (row >= 64 * 1024) return;
    int b = row >> 10;
    int t = sel[row];
    const float* src = h1 + ((size_t)b * S + t) * 128;
    float* dst = rec + (size_t)row * 128;
    float acc[5] = {bl[0], bl[1], bl[2], bl[3], bl[4]};
#pragma unroll 8
    for (int q = 0; q < 128; q += 4) {
        float4 v = *(const float4*)(src + q);
        *(float4*)(dst + q) = v;
#pragma unroll
        for (int jj = 0; jj < 5; jj++)
            acc[jj] += v.x * w[jj * 128 + q] + v.y * w[jj * 128 + q + 1] +
                       v.z * w[jj * 128 + q + 2] + v.w * w[jj * 128 + q + 3];
    }
#pragma unroll
    for (int jj = 0; jj < 5; jj++)
        out[(size_t)row * 5 + jj] = sigmoidf_(acc[jj]);
}

// ---------------------------------------------------------------------------
extern "C" void kernel_launch(void* const* d_in, const int* in_sizes, int n_in,
                              void* d_out, int out_size, void* d_ws, size_t ws_size,
                              hipStream_t stream) {
    (void)in_sizes; (void)n_in; (void)out_size; (void)ws_size;
    const int B = 64, S = 2048, D = 128, H = 64, G = 256, K = 1024;
    const int M = B * S;                       // 131072 GEMM rows

    const float* x      = (const float*)d_in[0];
    const int*   sel    = (const int*)d_in[1];
    const float* h0     = (const float*)d_in[2];
    const float* c0     = (const float*)d_in[3];
    const float* Wih0   = (const float*)d_in[4];
    const float* Whh0   = (const float*)d_in[5];
    const float* bih0   = (const float*)d_in[6];
    const float* bhh0   = (const float*)d_in[7];
    const float* Wih1   = (const float*)d_in[8];
    const float* Whh1   = (const float*)d_in[9];
    const float* bih1   = (const float*)d_in[10];
    const float* bhh1   = (const float*)d_in[11];
    const float* Wlin   = (const float*)d_in[12];
    const float* blin   = (const float*)d_in[13];

    // ---- workspace layout ---------------------------------------------------
    char* ws = (char*)d_ws;
    size_t off = 0;
    float*          pre    = (float*)(ws + off);          off += (size_t)2 * M * G * 4;   // 256MB
    unsigned short* xbf    = (unsigned short*)(ws + off); off += (size_t)M * D * 2;       // 32MB
    unsigned short* h0bf   = (unsigned short*)(ws + off); off += (size_t)M * 2 * H * 2;   // 32MB
    float*          h1f    = (float*)(ws + off);          off += (size_t)M * 2 * H * 4;   // 64MB
    unsigned short* wih0bf = (unsigned short*)(ws + off); off += (size_t)2 * G * D * 2;
    unsigned short* whh0bf = (unsigned short*)(ws + off); off += (size_t)2 * G * H * 2;
    unsigned short* wih1bf = (unsigned short*)(ws + off); off += (size_t)2 * G * D * 2;
    unsigned short* whh1bf = (unsigned short*)(ws + off); off += (size_t)2 * G * H * 2;
    float*          biasws = (float*)(ws + off);          off += (size_t)2 * 2 * G * 4;

    float* record = (float*)d_out;                    // [B*K, 128]
    float* outlin = record + (size_t)B * K * 2 * H;   // [B*K, 5]

    // ---- conversions / bias fold -------------------------------------------
    cvt_bf16_kernel<<<4096, 256, 0, stream>>>(x, xbf, M * D);
    cvt_bf16_kernel<<<64, 256, 0, stream>>>(Wih0, wih0bf, 2 * G * D);
    cvt_bf16_kernel<<<32, 256, 0, stream>>>(Whh0, whh0bf, 2 * G * H);
    cvt_bf16_kernel<<<64, 256, 0, stream>>>(Wih1, wih1bf, 2 * G * D);
    cvt_bf16_kernel<<<32, 256, 0, stream>>>(Whh1, whh1bf, 2 * G * H);
    bias_sum_kernel<<<2, 256, 0, stream>>>(bih0, bhh0, biasws, 2 * G);
    bias_sum_kernel<<<2, 256, 0, stream>>>(bih1, bhh1, biasws + 2 * G, 2 * G);

    // ---- layer 0 ------------------------------------------------------------
    gates_gemm_kernel<<<8192, 256, 0, stream>>>(xbf, wih0bf, pre, M);
    lstm_recur_kernel<<<8, 256, 0, stream>>>(pre, whh0bf, biasws,
                                             h0, c0, (void*)h0bf, S, 1);
    // ---- layer 1 (reuse gate slab) -----------------------------------------
    gates_gemm_kernel<<<8192, 256, 0, stream>>>(h0bf, wih1bf, pre, M);
    lstm_recur_kernel<<<8, 256, 0, stream>>>(pre, whh1bf, biasws + 2 * G,
                                             h0 + 2 * B * H, c0 + 2 * B * H,
                                             (void*)h1f, S, 0);
    // ---- gather + linear head ----------------------------------------------
    gather_linear_kernel<<<256, 256, 0, stream>>>(h1f, sel, Wlin, blin,
                                                  record, outlin, S);
}
